// BEM_50148038148677
// MI455X (gfx1250) — compile-verified, matrix operand order checked by reading, not run on hardware
//
#include <hip/hip_runtime.h>
#include <hip/hip_bf16.h>
#include <math.h>

// ---------------------------------------------------------------------------
// WMMA types & fragment helpers (CDNA5 / gfx1250, wave32)
// ---------------------------------------------------------------------------
typedef __attribute__((ext_vector_type(16))) __bf16 v16bf;
typedef __attribute__((ext_vector_type(8)))  float  v8f;

struct Bits32 { uint4 lo, hi; };           // 32 bytes == v16bf

__device__ __forceinline__ unsigned short f2bf(float f) {
  unsigned int u = __builtin_bit_cast(unsigned int, f);
  u = u + 0x7FFFu + ((u >> 16) & 1u);      // round-to-nearest-even
  return (unsigned short)(u >> 16);
}

// Unified 16x32 bf16 fragment load (A row-major, or B stored as column panels
// [16][K]).  Lane L covers row/col L%16; element i holds
// K = (i%8) + (i/8)*16 + (L/16)*8   (ISA 7.12.2).  Two 16B loads per lane.
__device__ __forceinline__ v16bf load_frag16(const unsigned short* base, int ld) {
  const int lane = threadIdx.x & 31;
  const int row  = lane & 15;
  const int half = lane >> 4;
  const unsigned short* p = base + (long long)row * ld + half * 8;
  Bits32 b;
  b.lo = *(const uint4*)(p);        // K = half*8 .. half*8+7
  b.hi = *(const uint4*)(p + 16);   // K = 16+half*8 ..
  return __builtin_bit_cast(v16bf, b);
}

#define WMMA_BF16(a, b, c) \
  __builtin_amdgcn_wmma_f32_16x16x32_bf16(false, (a), false, (b), (short)0, (c), false, false)

// ---------------------------------------------------------------------------
// sample GEMM: (512 x 256) * (256 x 65536) -> samp, written directly in the
// panel layout the r3d GEMM consumes:
//   SP[b][jp][col][kk]  with kk = c*32 + n,  j = jp*16+col,  m = n*2048 + j
// A = h_bf row-major; B = mask panels [4096][16][256].
// One wave computes a 32x64 C tile (2 M-tiles x 4 panels).  grid (1024, 16)
// ---------------------------------------------------------------------------
__global__ __launch_bounds__(32)
void wmma_gemm_sample(const unsigned short* __restrict__ A,
                      const unsigned short* __restrict__ Bp,
                      unsigned short* __restrict__ SP)
{
  const int p0  = blockIdx.x * 4;
  const int tm0 = blockIdx.y * 2;
  const unsigned short* Ab0 = A + (long long)tm0 * 16 * 256;
  const unsigned short* Ab1 = Ab0 + 16 * 256;
  v8f acc[2][4] = {};
  for (int k0 = 0; k0 < 256; k0 += 32) {
    if (k0 + 32 < 256) {
      __builtin_prefetch(Ab0 + k0 + 32, 0, 0);
      __builtin_prefetch(Ab1 + k0 + 32, 0, 0);
    }
    v16bf a0 = load_frag16(Ab0 + k0, 256);
    v16bf a1 = load_frag16(Ab1 + k0, 256);
#pragma unroll
    for (int pp = 0; pp < 4; ++pp) {
      v16bf b = load_frag16(Bp + (long long)(p0 + pp) * 4096 + k0, 256);
      acc[0][pp] = WMMA_BF16(a0, b, acc[0][pp]);
      acc[1][pp] = WMMA_BF16(a1, b, acc[1][pp]);
    }
  }
  const int lane = threadIdx.x & 31;
  const int col  = lane & 15, half = lane >> 4;
#pragma unroll
  for (int mt = 0; mt < 2; ++mt) {
#pragma unroll
    for (int pp = 0; pp < 4; ++pp) {
      int m  = (p0 + pp) * 16 + col;       // 0..65535
      int n  = m >> 11;                    // 0..31
      int jj = m & 2047;                   // j = t*8+w
      long long pbase = (long long)(jj >> 4) * 131072 + (long long)col * 8192 + n;
#pragma unroll
      for (int j = 0; j < 8; ++j) {
        int row = (tm0 + mt) * 16 + j + 8 * half;  // (b,c)
        int c   = row & 255;
        int bb  = row >> 8;
        SP[(long long)bb * 16777216 + pbase + (long long)c * 32] = f2bf(acc[mt][pp][j]);
      }
    }
  }
}

// ---------------------------------------------------------------------------
// r3d GEMM: Y[b,o,j] = sum_kk W3[o,kk] * SP[b][j/16][j%16][kk] + bias[o]
// M=512, K=8192, N=2048.  grid (32, 16, 2), 32x64 tile per wave.
// ---------------------------------------------------------------------------
__global__ __launch_bounds__(32)
void wmma_gemm_r3d(const unsigned short* __restrict__ W3,
                   const unsigned short* __restrict__ SP,
                   float* __restrict__ Y,
                   const float* __restrict__ bias)
{
  const int p0  = blockIdx.x * 4;
  const int tm0 = blockIdx.y * 2;
  const int b   = blockIdx.z;
  const unsigned short* Ab0 = W3 + (long long)tm0 * 16 * 8192;
  const unsigned short* Ab1 = Ab0 + 16 * 8192;
  const unsigned short* Bb  = SP + (long long)b * 16777216LL;
  v8f acc[2][4] = {};
  for (int k0 = 0; k0 < 8192; k0 += 32) {
    if (k0 + 32 < 8192) {
      __builtin_prefetch(Ab0 + k0 + 32, 0, 0);
      __builtin_prefetch(Ab1 + k0 + 32, 0, 0);
    }
    v16bf a0 = load_frag16(Ab0 + k0, 8192);
    v16bf a1 = load_frag16(Ab1 + k0, 8192);
#pragma unroll
    for (int pp = 0; pp < 4; ++pp) {
      v16bf bf = load_frag16(Bb + (long long)(p0 + pp) * 131072 + k0, 8192);
      acc[0][pp] = WMMA_BF16(a0, bf, acc[0][pp]);
      acc[1][pp] = WMMA_BF16(a1, bf, acc[1][pp]);
    }
  }
  const int lane = threadIdx.x & 31;
  const int col  = lane & 15, half = lane >> 4;
#pragma unroll
  for (int mt = 0; mt < 2; ++mt) {
#pragma unroll
    for (int pp = 0; pp < 4; ++pp) {
      int jcol = (p0 + pp) * 16 + col;
#pragma unroll
      for (int j = 0; j < 8; ++j) {
        int o = (tm0 + mt) * 16 + j + 8 * half;
        Y[(long long)b * 1048576 + (long long)o * 2048 + jcol] = acc[mt][pp][j] + bias[o];
      }
    }
  }
}

// ---------------------------------------------------------------------------
// r2d GEMM: R2[b,o,j] = sum_c W2[o,c] * YP[b][j/16][j%16][c] + bias[o]
// M=128, K=512, N=2048.  grid (32, 4, 2)
// ---------------------------------------------------------------------------
__global__ __launch_bounds__(32)
void wmma_gemm_r2d(const unsigned short* __restrict__ W2,
                   const unsigned short* __restrict__ YP,
                   float* __restrict__ R2,
                   const float* __restrict__ bias)
{
  const int p0  = blockIdx.x * 4;
  const int tm0 = blockIdx.y * 2;
  const int b   = blockIdx.z;
  const unsigned short* Ab0 = W2 + (long long)tm0 * 16 * 512;
  const unsigned short* Ab1 = Ab0 + 16 * 512;
  const unsigned short* Bb  = YP + (long long)b * 1048576LL;
  v8f acc[2][4] = {};
  for (int k0 = 0; k0 < 512; k0 += 32) {
    if (k0 + 32 < 512) {
      __builtin_prefetch(Ab0 + k0 + 32, 0, 0);
      __builtin_prefetch(Ab1 + k0 + 32, 0, 0);
    }
    v16bf a0 = load_frag16(Ab0 + k0, 512);
    v16bf a1 = load_frag16(Ab1 + k0, 512);
#pragma unroll
    for (int pp = 0; pp < 4; ++pp) {
      v16bf bf = load_frag16(Bb + (long long)(p0 + pp) * 8192 + k0, 512);
      acc[0][pp] = WMMA_BF16(a0, bf, acc[0][pp]);
      acc[1][pp] = WMMA_BF16(a1, bf, acc[1][pp]);
    }
  }
  const int lane = threadIdx.x & 31;
  const int col  = lane & 15, half = lane >> 4;
#pragma unroll
  for (int mt = 0; mt < 2; ++mt) {
#pragma unroll
    for (int pp = 0; pp < 4; ++pp) {
      int jcol = (p0 + pp) * 16 + col;
#pragma unroll
      for (int j = 0; j < 8; ++j) {
        int o = (tm0 + mt) * 16 + j + 8 * half;
        R2[(long long)b * 262144 + (long long)o * 2048 + jcol] = acc[mt][pp][j] + bias[o];
      }
    }
  }
}

// ---------------------------------------------------------------------------
// conv1: 1D conv k=3, same pad, 512->256 channels.  One thread per output.
// ---------------------------------------------------------------------------
__global__ void conv1_kernel(const float* __restrict__ x, const float* __restrict__ w,
                             const float* __restrict__ bias, float* __restrict__ out)
{
  int idx = blockIdx.x * blockDim.x + threadIdx.x;
  if (idx >= 2 * 256 * 256) return;
  int t = idx & 255;
  int o = (idx >> 8) & 255;
  int b = idx >> 16;
  const float* xb = x + (long long)b * 512 * 256;
  const float* wo = w + o * 512 * 3;
  float acc = bias[o];
  for (int i = 0; i < 512; ++i) {
    const float* xi = xb + i * 256;
    const float* wi = wo + i * 3;
    float s = xi[t] * wi[1];
    if (t > 0)   s += xi[t - 1] * wi[0];
    if (t < 255) s += xi[t + 1] * wi[2];
    acc += s;
  }
  out[idx] = acc;
}

// ---------------------------------------------------------------------------
// GroupNorm (biased var, eps=1e-5) + ReLU; optional bf16 side-output, either
// linear (bf_panel=0) or column-panel [j/16][j%16][C] layout (bf_panel=1).
// One block per (batch, group); in-place safe.
// ---------------------------------------------------------------------------
__global__ __launch_bounds__(256)
void gn_relu_kernel(const float* __restrict__ in, float* __restrict__ out,
                    unsigned short* __restrict__ out_bf,
                    const float* __restrict__ g, const float* __restrict__ bta,
                    int C, int S, int groups, int bf_panel)
{
  const int grp = blockIdx.x % groups;
  const int bat = blockIdx.x / groups;
  const int cpg = C / groups;
  const long long base = (long long)bat * C * S + (long long)grp * cpg * S;
  const int n = cpg * S;
  float s = 0.0f, s2 = 0.0f;
  for (int i = threadIdx.x; i < n; i += blockDim.x) {
    float v = in[base + i]; s += v; s2 += v * v;
  }
  __shared__ float sh[512];
  sh[threadIdx.x] = s; sh[256 + threadIdx.x] = s2;
  __syncthreads();
  for (int off = 128; off > 0; off >>= 1) {
    if ((int)threadIdx.x < off) {
      sh[threadIdx.x]       += sh[threadIdx.x + off];
      sh[256 + threadIdx.x] += sh[256 + threadIdx.x + off];
    }
    __syncthreads();
  }
  const float inv_n = 1.0f / (float)n;
  const float mean = sh[0] * inv_n;
  const float var  = sh[256] * inv_n - mean * mean;
  const float rstd = rsqrtf(var + 1e-5f);
  for (int i = threadIdx.x; i < n; i += blockDim.x) {
    int c = grp * cpg + i / S;
    float v = (in[base + i] - mean) * rstd * g[c] + bta[c];
    v = fmaxf(v, 0.0f);
    out[base + i] = v;
    if (out_bf) {
      long long oi;
      if (bf_panel) {
        int j = i % S;
        oi = (long long)bat * C * S + (long long)(j >> 4) * 16 * C
           + (long long)(j & 15) * C + c;
      } else {
        oi = base + i;
      }
      out_bf[oi] = f2bf(v);
    }
  }
}

// ---------------------------------------------------------------------------
// fp32 -> bf16 cast, linear (for row-major A operands)
// ---------------------------------------------------------------------------
__global__ void cast_bf16_kernel(const float* __restrict__ in,
                                 unsigned short* __restrict__ out, long long n)
{
  long long i = (long long)blockIdx.x * blockDim.x + threadIdx.x;
  if (i < n) out[i] = f2bf(in[i]);
}

// ---------------------------------------------------------------------------
// mask cast: row-major (256 x 65536) f32 -> column panels [4096][16][256] bf16
// ---------------------------------------------------------------------------
__global__ void cast_mask_panel_kernel(const float* __restrict__ in,
                                       unsigned short* __restrict__ out)
{
  long long i = (long long)blockIdx.x * blockDim.x + threadIdx.x;
  if (i >= 16777216LL) return;
  int tau = (int)(i >> 16);       // K row
  int m   = (int)(i & 65535);     // column
  out[(long long)(m >> 4) * 4096 + (long long)(m & 15) * 256 + tau] = f2bf(in[i]);
}

// ---------------------------------------------------------------------------
// head 3x3 conv, 128->128, pad 1 on (T=256, W=8) grid.
// ---------------------------------------------------------------------------
__global__ void conv3x3_kernel(const float* __restrict__ f, const float* __restrict__ w,
                               const float* __restrict__ bias, float* __restrict__ out)
{
  int idx = blockIdx.x * blockDim.x + threadIdx.x;
  if (idx >= 2 * 128 * 2048) return;
  int u = idx & 7;
  int t = (idx >> 3) & 255;
  int o = (idx >> 11) & 127;
  int b = idx >> 18;
  const float* fb = f + (long long)b * 128 * 2048;
  const float* wo = w + o * 128 * 9;
  float acc = bias[o];
  for (int i = 0; i < 128; ++i) {
    const float* fi = fb + i * 2048;
    const float* wi = wo + i * 9;
#pragma unroll
    for (int dh = 0; dh < 3; ++dh) {
      int tt = t + dh - 1;
      if (tt < 0 || tt > 255) continue;
#pragma unroll
      for (int dw = 0; dw < 3; ++dw) {
        int uu = u + dw - 1;
        if (uu < 0 || uu > 7) continue;
        acc += fi[tt * 8 + uu] * wi[dh * 3 + dw];
      }
    }
  }
  out[idx] = acc;
}

// ---------------------------------------------------------------------------
// head final: 1x1 conv 128->1 + sigmoid, write channel `chan` of (B,2,T,W)
// ---------------------------------------------------------------------------
__global__ void head_final_kernel(const float* __restrict__ t1, const float* __restrict__ w,
                                  const float* __restrict__ bias, float* __restrict__ out,
                                  int chan)
{
  int idx = blockIdx.x * blockDim.x + threadIdx.x;
  if (idx >= 2 * 2048) return;
  int j = idx & 2047;
  int b = idx >> 11;
  const float* tb = t1 + (long long)b * 128 * 2048 + j;
  float acc = bias[0];
  for (int c = 0; c < 128; ++c) acc += tb[(long long)c * 2048] * w[c];
  out[(long long)b * 4096 + (long long)chan * 2048 + j] = 1.0f / (1.0f + expf(-acc));
}

// ---------------------------------------------------------------------------
// Launcher
// ---------------------------------------------------------------------------
extern "C" void kernel_launch(void* const* d_in, const int* in_sizes, int n_in,
                              void* d_out, int out_size, void* d_ws, size_t ws_size,
                              hipStream_t stream)
{
  const float* x     = (const float*)d_in[0];
  const float* mask  = (const float*)d_in[1];
  const float* c1_w  = (const float*)d_in[2];
  const float* c1_b  = (const float*)d_in[3];
  const float* gn1_g = (const float*)d_in[4];
  const float* gn1_b = (const float*)d_in[5];
  const float* r3d_w = (const float*)d_in[6];
  const float* r3d_b = (const float*)d_in[7];
  const float* gn3_g = (const float*)d_in[8];
  const float* gn3_b = (const float*)d_in[9];
  const float* r2d_w = (const float*)d_in[10];
  const float* r2d_b = (const float*)d_in[11];
  const float* gn2_g = (const float*)d_in[12];
  const float* gn2_b = (const float*)d_in[13];
  const float* s1_w  = (const float*)d_in[14];
  const float* s1_b  = (const float*)d_in[15];
  const float* sgn_g = (const float*)d_in[16];
  const float* sgn_b = (const float*)d_in[17];
  const float* s2_w  = (const float*)d_in[18];
  const float* s2_b  = (const float*)d_in[19];
  const float* e1_w  = (const float*)d_in[20];
  const float* e1_b  = (const float*)d_in[21];
  const float* egn_g = (const float*)d_in[22];
  const float* egn_b = (const float*)d_in[23];
  const float* e2_w  = (const float*)d_in[24];
  const float* e2_b  = (const float*)d_in[25];
  float* out = (float*)d_out;

  char* ws = (char*)d_ws;
  size_t off = 0;
  auto alloc = [&](size_t bytes) -> void* {
    off = (off + 255) & ~(size_t)255;
    void* p = ws + off;
    off += bytes;
    return p;
  };

  float*          h      = (float*)alloc(131072ULL * 4);            // (2,256,256)
  unsigned short* h_bf   = (unsigned short*)alloc(131072ULL * 2);
  unsigned short* mask_p = (unsigned short*)alloc(16777216ULL * 2); // [4096][16][256]
  unsigned short* w3_bf  = (unsigned short*)alloc(4194304ULL * 2);
  unsigned short* w2_bf  = (unsigned short*)alloc(65536ULL * 2);
  unsigned short* samp_p = (unsigned short*)alloc(33554432ULL * 2); // [2][128][16][8192]
  float*          y      = (float*)alloc(2097152ULL * 4);           // (2,512,2048)
  unsigned short* y_p    = (unsigned short*)alloc(2097152ULL * 2);  // [2][128][16][512]
  float*          r2     = (float*)alloc(524288ULL * 4);            // (2,128,2048)
  float*          t1     = (float*)alloc(524288ULL * 4);

  // 1) conv1 + GN + ReLU (also emit bf16 h, linear row-major: it is A)
  conv1_kernel<<<512, 256, 0, stream>>>(x, c1_w, c1_b, h);
  gn_relu_kernel<<<64, 256, 0, stream>>>(h, h, h_bf, gn1_g, gn1_b, 256, 256, 32, 0);

  // 2) one-time bf16 casts of static operands
  cast_mask_panel_kernel<<<65536, 256, 0, stream>>>(mask, mask_p);
  cast_bf16_kernel<<<16384, 256, 0, stream>>>(r3d_w, w3_bf, 4194304LL);
  cast_bf16_kernel<<<256,   256, 0, stream>>>(r2d_w, w2_bf, 65536LL);

  // 3) sample GEMM: (512 x 256) * (256 x 65536) -> samp panels (bf16)
  wmma_gemm_sample<<<dim3(1024, 16, 1), 32, 0, stream>>>(h_bf, mask_p, samp_p);

  // 4) r3d GEMM: (512 x 8192) * (8192 x 2048) per batch -> y fp32 (+bias)
  wmma_gemm_r3d<<<dim3(32, 16, 2), 32, 0, stream>>>(w3_bf, samp_p, y, r3d_b);
  gn_relu_kernel<<<64, 256, 0, stream>>>(y, y, y_p, gn3_g, gn3_b, 512, 2048, 32, 1);

  // 5) r2d 1x1 GEMM: (128 x 512) * (512 x 2048) per batch -> r2 fp32 (+bias)
  wmma_gemm_r2d<<<dim3(32, 4, 2), 32, 0, stream>>>(w2_bf, y_p, r2, r2d_b);
  gn_relu_kernel<<<64, 256, 0, stream>>>(r2, r2, nullptr, gn2_g, gn2_b, 128, 2048, 32, 0);

  // 6) start head
  conv3x3_kernel<<<2048, 256, 0, stream>>>(r2, s1_w, s1_b, t1);
  gn_relu_kernel<<<64, 256, 0, stream>>>(t1, t1, nullptr, sgn_g, sgn_b, 128, 2048, 32, 0);
  head_final_kernel<<<16, 256, 0, stream>>>(t1, s2_w, s2_b, out, 0);

  // 7) end head
  conv3x3_kernel<<<2048, 256, 0, stream>>>(r2, e1_w, e1_b, t1);
  gn_relu_kernel<<<64, 256, 0, stream>>>(t1, t1, nullptr, egn_g, egn_b, 128, 2048, 32, 0);
  head_final_kernel<<<16, 256, 0, stream>>>(t1, e2_w, e2_b, out, 1);
}
